// DissipativeThetaRINN_79912161509968
// MI455X (gfx1250) — matrix operand order repaired and checked by reading, not compiled, not run on hardware
//
#include <hip/hip_runtime.h>
#include <hip/hip_bf16.h>
#include <stdint.h>

// ---------------- problem constants ----------------
#define B_TOT 1024
#define T_LEN 512
#define S_DIM 16
#define N_DIM 128
#define I_DIM 32
#define O_DIM 8
#define FPIT  30
#define DT_C  0.01f
#define OUTW  17          // 8 means + 8 stds + 1 value
#define VH    64          // value MLP hidden

// ---------------- WMMA types ----------------
typedef __attribute__((ext_vector_type(16))) __bf16 v16bf;
typedef __attribute__((ext_vector_type(8)))  float  v8f;

union V16 { v16bf v; unsigned u[8]; uint4 q[2]; };

__device__ __forceinline__ unsigned f2bf(float f) {
  unsigned u = __float_as_uint(f);
  return (u + 0x7FFFu + ((u >> 16) & 1u)) >> 16;           // RNE f32->bf16
}
// Truncating bf16x2 pack: single v_perm_b32 (top halves of a->lo, b->hi).
__device__ __forceinline__ unsigned pk2t(float a, float b) {
  return __builtin_amdgcn_perm(__float_as_uint(b), __float_as_uint(a),
                               0x07060302u);
}
__device__ __forceinline__ float fast_rcp(float x) {
#if __has_builtin(__builtin_amdgcn_rcpf)
  return __builtin_amdgcn_rcpf(x);
#else
  return 1.0f / x;
#endif
}
__device__ __forceinline__ float ftanh(float x) {
#if __has_builtin(__builtin_amdgcn_tanhf)
  return __builtin_amdgcn_tanhf(x);            // V_TANH_F32 (CDNA5 trans op)
#elif __has_builtin(__builtin_amdgcn_tanh_f32)
  return __builtin_amdgcn_tanh_f32(x);
#else
  float e = exp2f(2.88539008177793f * x);
  return 1.0f - 2.0f * fast_rcp(e + 1.0f);
#endif
}
__device__ __forceinline__ v8f wmma_bf16(v16bf a, v16bf b, v8f c) {
  return __builtin_amdgcn_wmma_f32_16x16x32_bf16(false, a, false, b, (short)0, c,
                                                 false, false);
}
__device__ __forceinline__ v8f zero8() {
  v8f z = {0.f, 0.f, 0.f, 0.f, 0.f, 0.f, 0.f, 0.f};
  return z;
}

// ---------------------------------------------------------------------------
// Weight operand images in d_ws (bf16), in WMMA A-operand per-lane layout:
//   element index within image: ((tc*32 + lane)*16 + e)
//   tc layout: [0,32)  dvw   (8 tiles x 4 chunks)   A[m][k] = Dvw_T[k][m]
//              [32,48) wbias (8 tiles x 2 chunks)   A[m][k] = [Cv_T;Dvy_T;0][k][m]
//              [48,50) wu    (1 tile  x 2 chunks)   A[m][k] = [Cu_T;Duy_T;0][k][m], m<8
//              [50,54) duw   (1 tile  x 4 chunks)   A[m][k] = Duw_T[k][m], m<8
//              [54,56) wx    (1 tile  x 2 chunks)   A[m][k] = [A_T;By_T;0][k][m]
//              [56,60) bw    (1 tile  x 4 chunks)   A[m][k] = Bw_T[k][m]
// A layout (16-bit, 16x32): lane: m = lane&15, half = lane>>4
//   element e: k = 32*chunk + e + 16*half
// ---------------------------------------------------------------------------
#define IMG_ELEMS (60 * 512)

__global__ void prep_images(const float* A_T, const float* Bw_T, const float* By_T,
                            const float* Cv_T, const float* Dvw_T, const float* Dvy_T,
                            const float* Cu_T, const float* Duw_T, const float* Duy_T,
                            unsigned short* img) {
  int g = blockIdx.x * blockDim.x + threadIdx.x;
  if (g >= IMG_ELEMS) return;
  int e = g & 15, lane = (g >> 4) & 31, tc = g >> 9;
  int half = lane >> 4, j = e >> 1, h = e & 1;
  int kk = e + 16 * half;   // k within 32-chunk
  int mlane = lane & 15;
  float val = 0.f;
  if (tc < 32) {                               // dvw
    int tile = tc >> 2, chunk = tc & 3;
    int m = 16 * tile + mlane, k = 32 * chunk + kk;
    val = Dvw_T[k * N_DIM + m];
  } else if (tc < 48) {                        // wbias
    int t2 = tc - 32, tile = t2 >> 1, chunk = t2 & 1;
    int m = 16 * tile + mlane, k = 32 * chunk + kk;
    if (k < 16)       val = Cv_T[k * N_DIM + m];
    else if (k < 48)  val = Dvy_T[(k - 16) * N_DIM + m];
  } else if (tc < 50) {                        // wu
    int chunk = tc - 48, m = mlane, k = 32 * chunk + kk;
    if (m < O_DIM) {
      if (k < 16)      val = Cu_T[k * O_DIM + m];
      else if (k < 48) val = Duy_T[(k - 16) * O_DIM + m];
    }
  } else if (tc < 54) {                        // duw
    int chunk = tc - 50, m = mlane, k = 32 * chunk + kk;
    if (m < O_DIM) val = Duw_T[k * O_DIM + m];
  } else if (tc < 56) {                        // wx
    int chunk = tc - 54, m = mlane, k = 32 * chunk + kk;
    if (k < 16)      val = A_T[k * S_DIM + m];
    else if (k < 48) val = By_T[(k - 16) * S_DIM + m];
  } else {                                     // bw
    int chunk = tc - 56, m = mlane, k = 32 * chunk + kk;
    val = Bw_T[k * S_DIM + m];
  }
  img[g] = (unsigned short)f2bf(val);
}

__device__ __forceinline__ v16bf load_tc(const unsigned short* img, int tc, int lane) {
  const uint4* p = (const uint4*)(img + (size_t)tc * 512 + (size_t)lane * 16);
  V16 t;
  t.q[0] = p[0];
  t.q[1] = p[1];
  return t.v;
}

// ---------------------------------------------------------------------------
// Controller: one wave (32 lanes) per 16 batch rows, runs full T scan.
// Everything stored TRANSPOSED: tiles are [feature x batch] so that WMMA D
// output maps back to the next B operand with only a shfl_xor(16).
//   lane: n = lane&15 (batch col), half = lane>>4
//   D tile: vgpr r -> feature m = r + 8*half (within 16-row tile)
//   B chunk c (K in [32c,32c+32)): lane holds col n; elem e: k = 32c + e + 16*half
// D tile t, vgpr r, lane (n+16*halfD)  ==>  chunk c=t>>1, lane (n+16*(t&1)),
//     elem e = r + 8*halfD
// ---------------------------------------------------------------------------
__global__ void __launch_bounds__(32, 1)
controller_kernel(const float* __restrict__ obs, const float* __restrict__ x0,
                  const unsigned short* __restrict__ img, float* __restrict__ out) {
  const int lane = threadIdx.x & 31;
  const int wid  = blockIdx.x;          // 0..63 -> batch tile
  const int b0   = wid * 16;
  const int n    = lane & 15;
  const int half = lane >> 4;

  // ---- register-resident weight operand images ----
  v16bf dvw[8][4];
#pragma unroll
  for (int t8 = 0; t8 < 8; ++t8)
#pragma unroll
    for (int c = 0; c < 4; ++c) dvw[t8][c] = load_tc(img, (t8 << 2) | c, lane);
  v16bf wb[8][2];
#pragma unroll
  for (int t8 = 0; t8 < 8; ++t8)
#pragma unroll
    for (int c = 0; c < 2; ++c) wb[t8][c] = load_tc(img, 32 + t8 * 2 + c, lane);
  v16bf wu[2];
#pragma unroll
  for (int c = 0; c < 2; ++c) wu[c] = load_tc(img, 48 + c, lane);
  v16bf duw[4];
#pragma unroll
  for (int c = 0; c < 4; ++c) duw[c] = load_tc(img, 50 + c, lane);
  v16bf wx[2];
#pragma unroll
  for (int c = 0; c < 2; ++c) wx[c] = load_tc(img, 54 + c, lane);
  v16bf bw[4];
#pragma unroll
  for (int c = 0; c < 4; ++c) bw[c] = load_tc(img, 56 + c, lane);

  // ---- recurrent state x^T tile (D layout: vgpr r -> m=r+8*half) ----
  v8f xT;
  {
    const float4* p = (const float4*)(x0 + (size_t)(b0 + n) * S_DIM + 8 * half);
    float4 a = p[0], b = p[1];
    xT[0] = a.x; xT[1] = a.y; xT[2] = a.z; xT[3] = a.w;
    xT[4] = b.x; xT[5] = b.y; xT[6] = b.z; xT[7] = b.w;
  }

  const float* obsbase = obs + (size_t)(b0 + n) * T_LEN * I_DIM + (half ? 0 : 16);
  float* outbase = out + (size_t)(b0 + n) * T_LEN * OUTW;

  for (int tt = 0; tt < T_LEN; ++tt) {
    // ---- build z^T = [x | y]^T (64x16, padded) as two B-operand chunks ----
    float ov[16];
    const float4* op = (const float4*)(obsbase + (size_t)tt * I_DIM);
#pragma unroll
    for (int i = 0; i < 4; ++i) {
      float4 f = op[i];
      ov[4 * i + 0] = f.x; ov[4 * i + 1] = f.y;
      ov[4 * i + 2] = f.z; ov[4 * i + 3] = f.w;
    }
    unsigned ow[8];
#pragma unroll
    for (int i = 0; i < 8; ++i) ow[i] = pk2t(ov[2 * i], ov[2 * i + 1]);
    unsigned xp[4], xs[4];
#pragma unroll
    for (int i = 0; i < 4; ++i) xp[i] = pk2t(xT[2 * i], xT[2 * i + 1]);
#pragma unroll
    for (int i = 0; i < 4; ++i) xs[i] = __shfl_xor(xp[i], 16);
    V16 z0, z1;
#pragma unroll
    for (int i = 0; i < 4; ++i) {
      z0.u[i]     = half ? ow[i]     : xp[i];   // k<16 : x rows (half0 lanes)
      z0.u[4 + i] = half ? ow[4 + i] : xs[i];   // half1 lanes: y rows 0..15
    }
#pragma unroll
    for (int i = 0; i < 8; ++i) z1.u[i] = half ? 0u : ow[i];  // y rows 16..31 / pad

    // ---- bias^T tiles = W_bias^T @ z^T  (8 tiles x 2 k-chunks) ----
    v8f bias[8];
#pragma unroll
    for (int t8 = 0; t8 < 8; ++t8) {
      v8f d = zero8();
      d = wmma_bf16(wb[t8][0], z0.v, d);
      d = wmma_bf16(wb[t8][1], z1.v, d);
      bias[t8] = d;
    }

    // ---- fixed-point: W^T = tanh(bias^T + Dvw @ W^T), 30 iters ----
    V16 Wc[4];
#pragma unroll
    for (int c = 0; c < 4; ++c)
#pragma unroll
      for (int i = 0; i < 8; ++i) Wc[c].u[i] = 0u;

#pragma unroll 1
    for (int it = 0; it < FPIT; ++it) {
      unsigned q[8][4];
#pragma unroll
      for (int t8 = 0; t8 < 8; ++t8) {
        v8f d = bias[t8];
#pragma unroll
        for (int c = 0; c < 4; ++c) d = wmma_bf16(dvw[t8][c], Wc[c].v, d);
#pragma unroll
        for (int i = 0; i < 4; ++i)
          q[t8][i] = pk2t(ftanh(d[2 * i]), ftanh(d[2 * i + 1]));
      }
      // repack D tiles -> next-iteration B chunks (single shfl per word-pair):
      //   a = own tile 2c word, b = own tile 2c+1 word
      //   publish s2 = half ? a : b;  sh = shfl_xor(s2,16)
      //   word[i]   = half ? sh : a   (halfD=0 source of tile 2c+H)
      //   word[4+i] = half ? b  : sh  (halfD=1 source of tile 2c+H)
#pragma unroll
      for (int c = 0; c < 4; ++c) {
#pragma unroll
        for (int i = 0; i < 4; ++i) {
          unsigned a = q[2 * c][i];
          unsigned b = q[2 * c + 1][i];
          unsigned s2 = half ? a : b;
          unsigned sh = __shfl_xor(s2, 16);
          Wc[c].u[i]     = half ? sh : a;
          Wc[c].u[4 + i] = half ? b : sh;
        }
      }
    }

    // ---- outputs: u^T and x-update, single tile each ----
    v8f uT = zero8();
    uT = wmma_bf16(wu[0], z0.v, uT);
    uT = wmma_bf16(wu[1], z1.v, uT);
#pragma unroll
    for (int c = 0; c < 4; ++c) uT = wmma_bf16(duw[c], Wc[c].v, uT);

    v8f xa = zero8();
    xa = wmma_bf16(wx[0], z0.v, xa);
    xa = wmma_bf16(wx[1], z1.v, xa);
#pragma unroll
    for (int c = 0; c < 4; ++c) xa = wmma_bf16(bw[c], Wc[c].v, xa);

#pragma unroll
    for (int r = 0; r < 8; ++r) xT[r] = fmaf(DT_C, xa[r], xT[r]);

    if (half == 0) {  // half0 lane n holds u[batch n][0..7]
      float* o = outbase + (size_t)tt * OUTW;
#pragma unroll
      for (int r = 0; r < 8; ++r) o[r] = uT[r];
    }
  }
}

// ---------------------------------------------------------------------------
// Value MLP (+ stds): one thread per (b,t); weights staged in LDS.
// ---------------------------------------------------------------------------
__global__ void __launch_bounds__(256)
value_kernel(const float* __restrict__ obs, const float* __restrict__ log_stds,
             const float* __restrict__ Vw1, const float* __restrict__ Vb1,
             const float* __restrict__ Vw2, const float* __restrict__ Vb2,
             const float* __restrict__ Vw3, const float* __restrict__ Vb3,
             float* __restrict__ out) {
  __shared__ float sW1[I_DIM * VH];
  __shared__ float sW2[VH * VH];
  __shared__ float sB1[VH], sB2[VH], sW3[VH], sLS[O_DIM], sB3[1];
  for (int i = threadIdx.x; i < I_DIM * VH; i += 256) sW1[i] = Vw1[i];
  for (int i = threadIdx.x; i < VH * VH; i += 256) sW2[i] = Vw2[i];
  if (threadIdx.x < VH) {
    sB1[threadIdx.x] = Vb1[threadIdx.x];
    sB2[threadIdx.x] = Vb2[threadIdx.x];
    sW3[threadIdx.x] = Vw3[threadIdx.x];
  }
  if (threadIdx.x < O_DIM) sLS[threadIdx.x] = log_stds[threadIdx.x];
  if (threadIdx.x == 0) sB3[0] = Vb3[0];
  __syncthreads();

  size_t g = (size_t)blockIdx.x * 256 + threadIdx.x;  // b*T + t
  if (g >= (size_t)B_TOT * T_LEN) return;
  const float* ob = obs + g * I_DIM;
  float x[I_DIM];
#pragma unroll
  for (int i = 0; i < I_DIM; ++i) x[i] = ob[i];

  float h1[VH];
#pragma unroll 8
  for (int o = 0; o < VH; ++o) {
    float s = sB1[o];
#pragma unroll
    for (int i = 0; i < I_DIM; ++i) s = fmaf(x[i], sW1[i * VH + o], s);
    h1[o] = ftanh(s);
  }
  float h2[VH];
#pragma unroll 4
  for (int o = 0; o < VH; ++o) {
    float s = sB2[o];
#pragma unroll
    for (int i = 0; i < VH; ++i) s = fmaf(h1[i], sW2[i * VH + o], s);
    h2[o] = ftanh(s);
  }
  float v = sB3[0];
#pragma unroll
  for (int o = 0; o < VH; ++o) v = fmaf(h2[o], sW3[o], v);

  float* o_ = out + g * OUTW;
#pragma unroll
  for (int i = 0; i < O_DIM; ++i) o_[8 + i] = sLS[i];
  o_[16] = v;
}

// ---------------------------------------------------------------------------
extern "C" void kernel_launch(void* const* d_in, const int* in_sizes, int n_in,
                              void* d_out, int out_size, void* d_ws, size_t ws_size,
                              hipStream_t stream) {
  const float* obs      = (const float*)d_in[0];
  const float* x0       = (const float*)d_in[1];
  const float* A_T      = (const float*)d_in[2];
  const float* Bw_T     = (const float*)d_in[3];
  const float* By_T     = (const float*)d_in[4];
  const float* Cv_T     = (const float*)d_in[5];
  const float* Dvw_T    = (const float*)d_in[6];
  const float* Dvy_T    = (const float*)d_in[7];
  const float* Cu_T     = (const float*)d_in[8];
  const float* Duw_T    = (const float*)d_in[9];
  const float* Duy_T    = (const float*)d_in[10];
  const float* log_stds = (const float*)d_in[11];
  const float* Vw1 = (const float*)d_in[12];
  const float* Vb1 = (const float*)d_in[13];
  const float* Vw2 = (const float*)d_in[14];
  const float* Vb2 = (const float*)d_in[15];
  const float* Vw3 = (const float*)d_in[16];
  const float* Vb3 = (const float*)d_in[17];
  float* out = (float*)d_out;
  unsigned short* img = (unsigned short*)d_ws;

  prep_images<<<(IMG_ELEMS + 255) / 256, 256, 0, stream>>>(
      A_T, Bw_T, By_T, Cv_T, Dvw_T, Dvy_T, Cu_T, Duw_T, Duy_T, img);

  controller_kernel<<<B_TOT / 16, 32, 0, stream>>>(obs, x0, img, out);

  value_kernel<<<(B_TOT * T_LEN + 255) / 256, 256, 0, stream>>>(
      obs, log_stds, Vw1, Vb1, Vw2, Vb2, Vw3, Vb3, out);
}
